// STAUpsample_47648367182191
// MI455X (gfx1250) — matrix operand order repaired and checked by reading, not compiled, not run on hardware
//
#include <hip/hip_runtime.h>
#include <hip/hip_bf16.h>

// ---------------------------------------------------------------------------
// STAUpsample for MI455X (gfx1250, wave32, WMMA)
// B=4, C=64, E=4, K=5, low-res 96x96, scale=2 -> hi-res 192x192
// ---------------------------------------------------------------------------

typedef __attribute__((ext_vector_type(16))) _Float16 v16h;
typedef __attribute__((ext_vector_type(8)))  _Float16 v8h;
typedef __attribute__((ext_vector_type(8)))  float    v8f;

union hv16 { v16h v; v8h h[2]; };

#define BB   4
#define CC   64
#define EE   4
#define HLR  96
#define WLR  96
#define HHR  192
#define WHR  192

#define AS1 __attribute__((address_space(1)))
#define AS3 __attribute__((address_space(3)))

#if defined(__AMDGCN__) && __has_builtin(__builtin_amdgcn_global_load_async_to_lds_b32) && \
    __has_builtin(__builtin_amdgcn_s_wait_asynccnt)
#define USE_ASYNC 1
#else
#define USE_ASYNC 0
#endif

// A-matrix (16x32 f16) element->K map (contiguous, matches B):
//   lanes 0-15 : elements 0..15 hold K=0..15; lanes 16-31: K=16..31
__device__ __forceinline__ int ka_map(int e, int lane) {
  return e + ((lane & 16) ? 16 : 0);
}
// B-matrix (32x16 f16): lanes 0-15 hold K=0..15, lanes 16-31 hold K=16..31 (contiguous)

__device__ __forceinline__ v8f wmma_f16(v16h a, v16h b, v8f c) {
  return __builtin_amdgcn_wmma_f32_16x16x32_f16(false, a, false, b,
                                                (short)0, c, false, false);
}

// ---------------------------------------------------------------------------
// Kernel 0: one-time f32 -> f16 weight conversion (kc_w 205KB, fus_w 16KB;
// both stay L2-resident for the whole dispatch -> B fragments load as b128
// of halves with no per-tile cvt).
// ---------------------------------------------------------------------------
__global__ void cvt_f16(const float* __restrict__ in, _Float16* __restrict__ out,
                        int n) {
  int i = blockIdx.x * 256 + threadIdx.x;
  if (i < n) out[i] = (_Float16)in[i];
}

// ---------------------------------------------------------------------------
// Kernel 1: fused  kernel_warp = leaky_relu(kc_w @ st_feat + kc_b)  (WMMA GEMM)
//           + spatially-varying 5x5 depthwise conv  (sta_feat)
// kernel_warp never touches HBM (236 MB saved each way): produced tile-by-tile
// into LDS and consumed immediately.
// One wave / block; block = 32 pixels (two 16-px M-tiles) of one (b, h) row,
// so every B fragment feeds 2 WMMAs.
// ---------------------------------------------------------------------------
__global__ __launch_bounds__(32) void k1_sta(const float* __restrict__ x,
                                             const float* __restrict__ stf,
                                             const _Float16* __restrict__ kcwh,
                                             const float* __restrict__ kcb,
                                             float* __restrict__ sta_out) {
  int blk = blockIdx.x;
  int wt = blk % (WLR / 32);
  int h  = (blk / (WLR / 32)) % HLR;
  int b  = blk / ((WLR / 32) * HLR);
  int w0 = wt * 32;
  int lane = threadIdx.x;
  int m = lane & 15;
  int koff  = (lane & 16) ? 16 : 0;
  int mbase = (lane & 16) ? 8 : 0;

  __shared__ float kwbuf[400 * 32];     // [o_local(400)][px(32)]  51.2 KB
  __shared__ float xtile[16 * 5 * 36];  // [c_local][dy][dx]       11.5 KB

  // A fragments: st_feat pixels [2 x (16 px x 64 K)], f32 -> f16 once.
  v16h aA[2][2];
#pragma unroll
  for (int t = 0; t < 2; ++t) {
    const float* sp = stf + (size_t)b * CC * HLR * WLR +
                      (size_t)h * WLR + w0 + t * 16 + m;
#pragma unroll
    for (int e = 0; e < 16; ++e) {
      int k = ka_map(e, lane);
      aA[t][0][e] = (_Float16)sp[(size_t)k * (HLR * WLR)];
      aA[t][1][e] = (_Float16)sp[(size_t)(k + 32) * (HLR * WLR)];
    }
  }

  for (int cb = 0; cb < 4; ++cb) {        // 16 x-channels per pass
    int obase = cb * 400;                 // o = c*25 + tap is contiguous

    // ---- GEMM: 25 N-tiles of 16 conv outputs, K=64, 2 M-tiles ----
    for (int nt = 0; nt < 25; ++nt) {
      int o0 = obase + nt * 16;
      const _Float16* wp = kcwh + (size_t)(o0 + m) * CC;  // row = column N
      __builtin_prefetch(wp + 16 * CC, 0, 0);             // next N-tile's row
      hv16 b0, b1;
      b0.h[0] = *(const v8h*)(wp + koff);
      b0.h[1] = *(const v8h*)(wp + koff + 8);
      b1.h[0] = *(const v8h*)(wp + 32 + koff);
      b1.h[1] = *(const v8h*)(wp + 32 + koff + 8);
      v8f c0 = {}, c1 = {};
      c0 = wmma_f16(aA[0][0], b0.v, c0);
      c0 = wmma_f16(aA[0][1], b1.v, c0);
      c1 = wmma_f16(aA[1][0], b0.v, c1);
      c1 = wmma_f16(aA[1][1], b1.v, c1);
      float bias = kcb[o0 + m];
#pragma unroll
      for (int r = 0; r < 8; ++r) {       // C: row px = r+mbase, col = m
        float v0 = c0[r] + bias;
        float v1 = c1[r] + bias;
        v0 = v0 > 0.f ? v0 : 0.1f * v0;   // leaky_relu(0.1)
        v1 = v1 > 0.f ? v1 : 0.1f * v1;
        kwbuf[(nt * 16 + m) * 32 + mbase + r]      = v0;
        kwbuf[(nt * 16 + m) * 32 + 16 + mbase + r] = v1;
      }
    }

    // ---- stage x tile (replicate padding): [16c][5][36] ----
    const float* xb = x + (size_t)b * CC * HLR * WLR +
                      (size_t)(cb * 16) * HLR * WLR;
    for (int t = lane; t < 16 * 5 * 36; t += 32) {
      int dx = t % 36;
      int dy = (t / 36) % 5;
      int cl = t / 180;
      int yy = h + dy - 2;  yy = yy < 0 ? 0 : (yy > HLR - 1 ? HLR - 1 : yy);
      int xx = w0 + dx - 2; xx = xx < 0 ? 0 : (xx > WLR - 1 ? WLR - 1 : xx);
      const float* gp = xb + (size_t)cl * HLR * WLR + (size_t)yy * WLR + xx;
#if USE_ASYNC
      __builtin_amdgcn_global_load_async_to_lds_b32((AS1 int*)gp,
                                                    (AS3 int*)&xtile[t], 0, 0);
#else
      xtile[t] = *gp;
#endif
    }
#if USE_ASYNC
    __builtin_amdgcn_s_wait_asynccnt(0);
#endif
    __syncthreads();

    // ---- depthwise 5x5 with per-pixel kernels from LDS ----
    for (int p = lane; p < 512; p += 32) {  // 16 channels x 32 pixels
      int px = p & 31;
      int cl = p >> 5;
      const float* kw = &kwbuf[cl * 25 * 32 + px];
      const float* xt = &xtile[cl * 180];
      float acc = 0.f;
#pragma unroll
      for (int dy = 0; dy < 5; ++dy)
#pragma unroll
        for (int dx = 0; dx < 5; ++dx)
          acc += kw[(dy * 5 + dx) * 32] * xt[dy * 36 + px + dx];
      sta_out[((size_t)b * CC + cb * 16 + cl) * HLR * WLR +
              (size_t)h * WLR + w0 + px] = acc;
    }
    __syncthreads();
  }
}

// ---------------------------------------------------------------------------
// Kernel 2: coordinate MLP (4 -> 64 -> 64) + heads (routing/offset/st_offset).
// Layer 1 (K=4) computed directly into A fragments; layer 2 via WMMA.
// ---------------------------------------------------------------------------
__global__ __launch_bounds__(32) void k2_emb(const float* __restrict__ b1w,
                                             const float* __restrict__ b1b,
                                             const float* __restrict__ b2w,
                                             const float* __restrict__ b2b,
                                             const float* __restrict__ rw,
                                             const float* __restrict__ rb,
                                             const float* __restrict__ ow,
                                             const float* __restrict__ ob,
                                             const float* __restrict__ sow,
                                             const float* __restrict__ sob,
                                             float* __restrict__ routing,
                                             float* __restrict__ offset,
                                             float* __restrict__ stoff) {
  int blk = blockIdx.x;
  int xt = blk % (WHR / 16);
  int y  = blk / (WHR / 16);
  int x0 = xt * 16;
  int lane = threadIdx.x;
  int m = lane & 15;
  int px = x0 + m;
  int koff  = (lane & 16) ? 16 : 0;
  int mbase = (lane & 16) ? 8 : 0;

  __shared__ float embs[16 * 64];  // [px][64]

  float chf = (y + 0.5f) * 0.5f;                  // scale = 2
  float cwf = (px + 0.5f) * 0.5f;
  float inp0 = 0.5f, inp1 = 0.5f;
  float inp2 = chf - floorf(chf + 0.001f) - 0.5f;
  float inp3 = cwf - floorf(cwf + 0.001f) - 0.5f;

  v16h a0, a1;
#pragma unroll
  for (int e = 0; e < 16; ++e) {
    int k0 = ka_map(e, lane);
    int k1 = k0 + 32;
    float v0 = b1w[k0 * 4 + 0] * inp0 + b1w[k0 * 4 + 1] * inp1 +
               b1w[k0 * 4 + 2] * inp2 + b1w[k0 * 4 + 3] * inp3 + b1b[k0];
    float v1 = b1w[k1 * 4 + 0] * inp0 + b1w[k1 * 4 + 1] * inp1 +
               b1w[k1 * 4 + 2] * inp2 + b1w[k1 * 4 + 3] * inp3 + b1b[k1];
    a0[e] = (_Float16)(v0 > 0.f ? v0 : 0.f);
    a1[e] = (_Float16)(v1 > 0.f ? v1 : 0.f);
  }

  for (int nt = 0; nt < 4; ++nt) {
    v16h bf0, bf1;
    const float* wp = b2w + (size_t)(nt * 16 + m) * 64;
#pragma unroll
    for (int e = 0; e < 16; ++e) {
      bf0[e] = (_Float16)wp[e + koff];
      bf1[e] = (_Float16)wp[e + koff + 32];
    }
    v8f c = {};
    c = wmma_f16(a0, bf0, c);
    c = wmma_f16(a1, bf1, c);
    float bias = b2b[nt * 16 + m];
#pragma unroll
    for (int r = 0; r < 8; ++r) {
      float v = c[r] + bias;
      embs[(mbase + r) * 64 + nt * 16 + m] = v > 0.f ? v : 0.f;
    }
  }
  __syncthreads();

  for (int t = lane; t < 128; t += 32) {
    int pj = t >> 3;
    int j  = t & 7;
    const float* w;
    float acc;
    if (j < 4)      { w = rw  + j * 64;       acc = rb[j];      }
    else if (j < 6) { w = ow  + (j - 4) * 64; acc = ob[j - 4];  }
    else            { w = sow + (j - 6) * 64; acc = sob[j - 6]; }
#pragma unroll 8
    for (int k = 0; k < 64; ++k) acc += w[k] * embs[pj * 64 + k];
    size_t pix = (size_t)y * WHR + x0 + pj;
    if (j < 4)      routing[(size_t)j * HHR * WHR + pix] = 1.f / (1.f + __expf(-acc));
    else if (j < 6) offset[(size_t)(j - 4) * HHR * WHR + pix] = acc;
    else            stoff[(size_t)(j - 6) * HHR * WHR + pix] = acc;
  }
}

// bilinear taps (zero padding outside; coords per reference formula)
__device__ __forceinline__ void make_taps(float sx, float sy, int* idx, float* wt) {
  float fx = floorf(sx), fy = floorf(sy);
  float wx1 = sx - fx, wy1 = sy - fy;
  float wx0 = 1.f - wx1, wy0 = 1.f - wy1;
  float wgt[4] = {wy0 * wx0, wy0 * wx1, wy1 * wx0, wy1 * wx1};
  int ix = (int)fx, iy = (int)fy;
#pragma unroll
  for (int t = 0; t < 4; ++t) {
    int xx = ix + (t & 1), yy = iy + (t >> 1);
    bool v = (xx >= 0) && (xx < WLR) && (yy >= 0) && (yy < HLR);
    int xc = xx < 0 ? 0 : (xx > WLR - 1 ? WLR - 1 : xx);
    int yc = yy < 0 ? 0 : (yy > HLR - 1 ? HLR - 1 : yy);
    idx[t] = yc * WLR + xc;
    wt[t]  = v ? wgt[t] : 0.f;
  }
}

// ---------------------------------------------------------------------------
// Kernel 3: grid-sample both streams, rank-8 mixture-of-experts bottleneck
// (f32 VALU; K=8 below WMMA shapes), then 128->64 fusion conv via WMMA with
// pre-converted f16 weights.
// ---------------------------------------------------------------------------
__global__ __launch_bounds__(32) void k3_main(const float* __restrict__ x,
                                              const float* __restrict__ sta,
                                              const float* __restrict__ wc,
                                              const float* __restrict__ we,
                                              const _Float16* __restrict__ fuswh,
                                              const float* __restrict__ fusb,
                                              const float* __restrict__ routing,
                                              const float* __restrict__ offs,
                                              const float* __restrict__ stoffs,
                                              float* __restrict__ out) {
  int blk = blockIdx.x;
  int xt = blk % (WHR / 16);
  int y  = blk / (WHR / 16);
  int x0 = xt * 16;
  int lane = threadIdx.x;
  int m = lane & 15;
  int X = x0 + m;
  size_t pix = (size_t)y * WHR + X;
  int chalf = (lane >> 4) * 32;
  int koff  = (lane & 16) ? 16 : 0;
  int mbase = (lane & 16) ? 8 : 0;

  __shared__ float fbuf[16 * 128]; // [px][0..63]=sta_hr, [64..127]=fea0/fea
  __shared__ float tbuf[16 * 8];
  __shared__ float rsh[16 * 4];

  for (int t = lane; t < 64; t += 32)
    rsh[t] = routing[(size_t)(t & 3) * HHR * WHR + (size_t)y * WHR + x0 + (t >> 2)];

  float sx  = (X + 0.5f) * 0.5f - 0.5f + offs[pix];
  float sy  = (y + 0.5f) * 0.5f - 0.5f + offs[(size_t)HHR * WHR + pix];
  float ssx = (X + 0.5f) * 0.5f - 0.5f + stoffs[pix];
  float ssy = (y + 0.5f) * 0.5f - 0.5f + stoffs[(size_t)HHR * WHR + pix];
  int sidx[4]; float swt[4];
  int tidx[4]; float twt[4];
  make_taps(sx, sy, sidx, swt);
  make_taps(ssx, ssy, tidx, twt);

  for (int b = 0; b < BB; ++b) {
    __syncthreads();

    const float* xb = x   + (size_t)b * CC * HLR * WLR;
    const float* sb = sta + (size_t)b * CC * HLR * WLR;
    for (int c2 = 0; c2 < 32; ++c2) {
      const float* cpx = xb + (size_t)(chalf + c2) * HLR * WLR;
      const float* cps = sb + (size_t)(chalf + c2) * HLR * WLR;
      fbuf[m * 128 + 64 + chalf + c2] =
          swt[0] * cpx[sidx[0]] + swt[1] * cpx[sidx[1]] +
          swt[2] * cpx[sidx[2]] + swt[3] * cpx[sidx[3]];
      fbuf[m * 128 + chalf + c2] =
          twt[0] * cps[tidx[0]] + twt[1] * cps[tidx[1]] +
          twt[2] * cps[tidx[2]] + twt[3] * cps[tidx[3]];
    }
    __syncthreads();

    // t[px,o] = sum_e r_e * (wc[e,o,:] . fea0[px,:])
    for (int t = lane; t < 128; t += 32) {
      int po = t & 7, pj = t >> 3;
      const float* fp = &fbuf[pj * 128 + 64];
      float acc = 0.f;
#pragma unroll
      for (int e = 0; e < EE; ++e) {
        const float* wp = wc + (size_t)(e * 8 + po) * 64;
        float d = 0.f;
#pragma unroll 8
        for (int c2 = 0; c2 < 64; ++c2) d += wp[c2] * fp[c2];
        acc += rsh[pj * 4 + e] * d;
      }
      tbuf[pj * 8 + po] = acc;
    }
    __syncthreads();

    // fea[px,c] = fea0 + sum_e r_e * (we[e,c,:] . t[px,:])   (in place)
    for (int t = lane; t < 1024; t += 32) {
      int c2 = t & 63, pj = t >> 6;
      float acc = 0.f;
#pragma unroll
      for (int e = 0; e < EE; ++e) {
        const float* wp = we + (size_t)(e * 64 + c2) * 8;
        float d = 0.f;
#pragma unroll
        for (int o = 0; o < 8; ++o) d += wp[o] * tbuf[pj * 8 + o];
        acc += rsh[pj * 4 + e] * d;
      }
      fbuf[pj * 128 + 64 + c2] += acc;
    }
    __syncthreads();

    // fusion conv: [16px x 128] x [128 -> 64] via 16 WMMAs
    v16h a[4];
#pragma unroll
    for (int f = 0; f < 4; ++f)
#pragma unroll
      for (int e = 0; e < 16; ++e)
        a[f][e] = (_Float16)fbuf[m * 128 + f * 32 + ka_map(e, lane)];

    for (int nt = 0; nt < 4; ++nt) {
      v8f c = {};
#pragma unroll
      for (int f = 0; f < 4; ++f) {
        const _Float16* wp = fuswh + (size_t)(nt * 16 + m) * 128 + f * 32;
        hv16 bu;
        bu.h[0] = *(const v8h*)(wp + koff);
        bu.h[1] = *(const v8h*)(wp + koff + 8);
        c = wmma_f16(a[f], bu.v, c);
      }
      float bias = fusb[nt * 16 + m];
#pragma unroll
      for (int r = 0; r < 8; ++r)     // lane's 8 outputs are W-contiguous
        out[((size_t)(b * CC + nt * 16 + m) * HHR + y) * WHR + x0 + mbase + r] =
            c[r] + bias;
    }
  }
}

// ---------------------------------------------------------------------------
extern "C" void kernel_launch(void* const* d_in, const int* in_sizes, int n_in,
                              void* d_out, int out_size, void* d_ws, size_t ws_size,
                              hipStream_t stream) {
  const float* x    = (const float*)d_in[0];
  const float* stf  = (const float*)d_in[1];
  const float* kcw  = (const float*)d_in[2];
  const float* kcb  = (const float*)d_in[3];
  const float* wc   = (const float*)d_in[4];
  const float* we   = (const float*)d_in[5];
  const float* b1w  = (const float*)d_in[6];
  const float* b1b  = (const float*)d_in[7];
  const float* b2w  = (const float*)d_in[8];
  const float* b2b  = (const float*)d_in[9];
  const float* rw   = (const float*)d_in[10];
  const float* rb   = (const float*)d_in[11];
  const float* ow   = (const float*)d_in[12];
  const float* ob   = (const float*)d_in[13];
  const float* sow  = (const float*)d_in[14];
  const float* sob  = (const float*)d_in[15];
  const float* fusw = (const float*)d_in[16];
  const float* fusb = (const float*)d_in[17];
  // d_in[18] = scale (2) -- baked into the kernels.

  char* wsp = (char*)d_ws;
  float* sta     = (float*)wsp; wsp += (size_t)BB * CC * HLR * WLR * sizeof(float);
  float* routing = (float*)wsp; wsp += (size_t)EE * HHR * WHR * sizeof(float);
  float* offset  = (float*)wsp; wsp += (size_t)2 * HHR * WHR * sizeof(float);
  float* stoff   = (float*)wsp; wsp += (size_t)2 * HHR * WHR * sizeof(float);
  _Float16* kcwh  = (_Float16*)wsp; wsp += (size_t)1600 * 64 * sizeof(_Float16);
  _Float16* fuswh = (_Float16*)wsp; wsp += (size_t)64 * 128 * sizeof(_Float16);

  dim3 blk(32);
  cvt_f16<<<dim3((1600 * 64 + 255) / 256), dim3(256), 0, stream>>>(kcw, kcwh, 1600 * 64);
  cvt_f16<<<dim3((64 * 128 + 255) / 256), dim3(256), 0, stream>>>(fusw, fuswh, 64 * 128);

  k1_sta<<<dim3(BB * HLR * (WLR / 32)), blk, 0, stream>>>(x, stf, kcwh, kcb, sta);
  k2_emb<<<dim3(HHR * (WHR / 16)), blk, 0, stream>>>(
      b1w, b1b, b2w, b2b, rw, rb, ow, ob, sow, sob, routing, offset, stoff);
  k3_main<<<dim3(HHR * (WHR / 16)), blk, 0, stream>>>(
      x, sta, wc, we, fuswh, fusb, routing, offset, stoff, (float*)d_out);
}
